// Fp8Linear_57956288692383
// MI455X (gfx1250) — 8-bit matrix operand order repaired and checked by reading, compiled, not run on hardware
//
#include <hip/hip_runtime.h>
#include <hip/hip_bf16.h>
#include <stdint.h>


#define G 128
#define KB_MAX 32

typedef int   v16i __attribute__((ext_vector_type(16)));
typedef int   v4i  __attribute__((ext_vector_type(4)));
typedef float v8f  __attribute__((ext_vector_type(8)));

#if defined(__AMDGCN__) && __has_builtin(__builtin_amdgcn_tensor_load_to_lds)
#define FILL_TDM 1
#elif defined(__AMDGCN__) && __has_builtin(__builtin_amdgcn_global_load_async_to_lds_b128)
#define FILL_ASYNC 1
#else
#define FILL_SYNC 1
#endif


__device__ __forceinline__ unsigned f32_to_fp8_e4m3_sw(float x) {
    unsigned u    = __float_as_uint(x);
    unsigned sign = (u >> 31) << 7;
    float a = fabsf(x);
    if (!(a > 0.0f)) return sign;
    a = fminf(a, 448.0f);
    if (a < 0.0009765625f) return sign;
    unsigned au = __float_as_uint(a);
    int e = (int)(au >> 23) - 127;
    if (e < -6) e = -6;
    float quantum = __uint_as_float((unsigned)(e - 3 + 127) << 23);
    float r = rintf(a / quantum) * quantum;
    unsigned ru = __float_as_uint(r);
    int re = (int)(ru >> 23) - 127;
    unsigned code;
    if (re < -6) code = (unsigned)rintf(r * 512.0f);
    else         code = ((unsigned)(re + 7) << 3) | ((ru >> 20) & 7);
    if (code > 0x7Eu) code = 0x7Eu;
    return sign | code;
}

__device__ __forceinline__ unsigned pack4_fp8(float a, float b, float c, float d) {
#if __has_builtin(__builtin_amdgcn_cvt_pk_fp8_f32)
    int p = 0;
    p = __builtin_amdgcn_cvt_pk_fp8_f32(a, b, p, false);
    p = __builtin_amdgcn_cvt_pk_fp8_f32(c, d, p, true);
    return (unsigned)p;
#else
    return  f32_to_fp8_e4m3_sw(a)
         | (f32_to_fp8_e4m3_sw(b) << 8)
         | (f32_to_fp8_e4m3_sw(c) << 16)
         | (f32_to_fp8_e4m3_sw(d) << 24);
#endif
}

__global__ __launch_bounds__(256) void quant_act_kernel(
    const float* __restrict__ x, uint8_t* __restrict__ qx,
    float* __restrict__ sxT, int M, int K, int KB)
{
    int gid  = blockIdx.x * 8 + (threadIdx.x >> 5);
    int lane = threadIdx.x & 31;
    int m  = gid / KB;
    int kb = gid - m * KB;

    const float4* px = (const float4*)(x + (size_t)m * K + kb * G) + lane;
    float4 v = *px;

    float amax = fmaxf(fmaxf(fabsf(v.x), fabsf(v.y)), fmaxf(fabsf(v.z), fabsf(v.w)));
#pragma unroll
    for (int off = 16; off > 0; off >>= 1)
        amax = fmaxf(amax, __shfl_xor(amax, off, 32));

    float s   = fmaxf(amax, 1e-12f) * (1.0f / 448.0f);
    float inv = 1.0f / s;

    unsigned packed = pack4_fp8(v.x * inv, v.y * inv, v.z * inv, v.w * inv);
    const int nw = ((lane >> 2) & 1) * 16 + (lane >> 3) * 4 + (lane & 3);
    *(unsigned*)(qx + (size_t)m * K + kb * G + nw * 4) = packed;
    if (lane == 0) sxT[(size_t)kb * M + m] = s;
}

__global__ __launch_bounds__(256) void quant_wt_kernel(
    const float* __restrict__ wq, uint8_t* __restrict__ qw8, size_t total4)
{
    size_t i = ((size_t)blockIdx.x * 256 + threadIdx.x);
    if (i >= total4) return;
    const float4 v = ((const float4*)wq)[i];
    const unsigned p = pack4_fp8(v.x, v.y, v.z, v.w);
    const size_t gbase = (i >> 5) << 5;
    const int w = (int)(i & 31);
    const int nw = ((w >> 2) & 1) * 16 + (w >> 3) * 4 + (w & 3);
    ((unsigned*)qw8)[gbase + nw] = p;
}


#if defined(FILL_TDM)
typedef unsigned int u32x4 __attribute__((ext_vector_type(4)));
typedef int          i32x4t __attribute__((ext_vector_type(4)));
typedef int          i32x8t __attribute__((ext_vector_type(8)));

__device__ __forceinline__ void tdm_load_2d(const void* gsrc, unsigned lds_off,
                                            unsigned dsz_code,
                                            unsigned tile_d0, unsigned tile_d1,
                                            unsigned tensor_d0, unsigned stride_d0)
{
    unsigned long long ga = (unsigned long long)(uintptr_t)gsrc;
    u32x4 g0 = {0u, 0u, 0u, 0u};
    g0[0] = 1u;
    g0[1] = lds_off;
    g0[2] = (unsigned)ga;
    g0[3] = (unsigned)((ga >> 32) & 0x01FFFFFFu) | 0x80000000u;

    i32x8t g1 = {0, 0, 0, 0, 0, 0, 0, 0};
    const unsigned td0 = tensor_d0;
    const unsigned td1 = 1u << 20;
    g1[0] = (int)(dsz_code << 16);
    g1[1] = (int)((td0 & 0xFFFFu) << 16);
    g1[2] = (int)(((td0 >> 16) & 0xFFFFu) | ((td1 & 0xFFFFu) << 16));
    g1[3] = (int)(((td1 >> 16) & 0xFFFFu) | ((tile_d0 & 0xFFFFu) << 16));
    g1[4] = (int)(tile_d1 & 0xFFFFu);
    g1[5] = (int)stride_d0;

    i32x4t gz = {0, 0, 0, 0};
#if __clang_major__ >= 23
    i32x8t gz8 = {0, 0, 0, 0, 0, 0, 0, 0};
    __builtin_amdgcn_tensor_load_to_lds(g0, g1, gz, gz, gz8, 0);
#else
    __builtin_amdgcn_tensor_load_to_lds(g0, g1, gz, gz, 0);
#endif
}
__device__ __forceinline__ unsigned lds_byte_off(const void* p) {
    return (unsigned)(uintptr_t)p;
}
#elif defined(FILL_ASYNC)
typedef __attribute__((address_space(3))) uint8_t        lds_u8_t;
typedef __attribute__((address_space(1))) const uint8_t  gbl_u8_t;
__device__ __forceinline__ void async_cp16(const void* g, void* l) {
    __builtin_amdgcn_global_load_async_to_lds_b128((gbl_u8_t*)g, (lds_u8_t*)l, 0, 0);
}
__device__ __forceinline__ void async_wait0() {
#if __has_builtin(__builtin_amdgcn_s_wait_asynccnt)
    __builtin_amdgcn_s_wait_asynccnt(0);
#else
    asm volatile("s_wait_asynccnt 0x0" ::: "memory");
#endif
}
#endif

__device__ __forceinline__ v16i lds_frag64(const uint8_t* p) {
    v16i f;
#pragma unroll
    for (int q = 0; q < 4; ++q) {
        v4i d = *(const v4i*)(p + 16 * q);
        f[4 * q]     = d.x;
        f[4 * q + 1] = d.y;
        f[4 * q + 2] = d.z;
        f[4 * q + 3] = d.w;
    }
    return f;
}

__global__ __launch_bounds__(256) void gemm_fp8_kernel(
    const uint8_t* __restrict__ qx, const float* __restrict__ sxT,
    const uint8_t* __restrict__ qw8, const float* __restrict__ wscale,
    const float* __restrict__ bias, float* __restrict__ out,
    int M, int N, int K, int KB)
{
    __shared__ __align__(16) uint8_t lA[2][G * G];
    __shared__ __align__(16) uint8_t lB[2][G * G];
    __shared__ __align__(16) float   lS[KB_MAX * G];

    const int tid  = threadIdx.x;
    const int wave = tid >> 5;
    const int lane = tid & 31;
    const int hi   = (lane >> 4) & 1;
    const int lan  = lane & 15;
    const int wm   = wave & 3;
    const int wn   = wave >> 2;

    const int n0 = blockIdx.x * G;
    const int m0 = blockIdx.y * G;
    const int nb = blockIdx.x;

    const v8f vzero = (v8f){};
    v8f acc[2][4];
#pragma unroll
    for (int u = 0; u < 2; ++u)
#pragma unroll
        for (int t = 0; t < 4; ++t) acc[u][t] = vzero;

    const int arow0 = (wm * 32 + lan) * G + hi * 64;
    const int brow0 = (wn * 64 + lan) * G + hi * 64;
    const int srow0 = wm * 32 + 8 * hi;

#if defined(FILL_TDM)
    if (tid < 32) {
        tdm_load_2d(qx  + (size_t)m0 * K, lds_byte_off(&lA[0][0]),
                    0, G, G, (unsigned)K, (unsigned)K);
        tdm_load_2d(qw8 + (size_t)n0 * K, lds_byte_off(&lB[0][0]),
                    0, G, G, (unsigned)K, (unsigned)K);
        tdm_load_2d(sxT + m0, lds_byte_off(&lS[0]),
                    2, G, (unsigned)KB, (unsigned)M, (unsigned)M);
    }
    __builtin_amdgcn_s_wait_tensorcnt(0);
    __syncthreads();
#elif defined(FILL_ASYNC)
    {
        const int frow = tid >> 1, fcol = (tid & 1) * 64;
        const uint8_t* gA = qx  + (size_t)(m0 + frow) * K + fcol;
        const uint8_t* gB = qw8 + (size_t)(n0 + frow) * K + fcol;
        uint8_t* sA = &lA[0][frow * G + fcol];
        uint8_t* sB = &lB[0][frow * G + fcol];
#pragma unroll
        for (int q = 0; q < 4; ++q) {
            async_cp16(gA + 16 * q, sA + 16 * q);
            async_cp16(gB + 16 * q, sB + 16 * q);
        }
        const int srow = tid >> 3, scol = (tid & 7) * 16;
        if (srow < KB) {
            const float* gs = sxT + (size_t)srow * M + m0 + scol;
            float* ls = &lS[srow * G + scol];
#pragma unroll
            for (int q = 0; q < 4; ++q) async_cp16(gs + 4 * q, ls + 4 * q);
        }
        async_wait0();
    }
    __syncthreads();
#else
    {
        const int frow = tid >> 1, fcol = (tid & 1) * 64;
        const v4i* gA = (const v4i*)(qx  + (size_t)(m0 + frow) * K + fcol);
        const v4i* gB = (const v4i*)(qw8 + (size_t)(n0 + frow) * K + fcol);
        v4i* sA = (v4i*)&lA[0][frow * G + fcol];
        v4i* sB = (v4i*)&lB[0][frow * G + fcol];
#pragma unroll
        for (int q = 0; q < 4; ++q) { sA[q] = gA[q]; sB[q] = gB[q]; }
        const int srow = tid >> 3, scol = (tid & 7) * 16;
        if (srow < KB) {
            const float4* gs = (const float4*)(sxT + (size_t)srow * M + m0 + scol);
            float4* ls = (float4*)&lS[srow * G + scol];
#pragma unroll
            for (int q = 0; q < 4; ++q) ls[q] = gs[q];
        }
    }
    __syncthreads();
#endif

    for (int kb = 0; kb < KB; ++kb) {
        const int cur = kb & 1;
        const uint8_t* curA = &lA[cur][0];
        const uint8_t* curB = &lB[cur][0];
        const bool have_next = (kb + 1 < KB);

#if defined(FILL_TDM)
        if (have_next && tid < 32) {
            const int nxt = cur ^ 1;
            tdm_load_2d(qx  + (size_t)m0 * K + (size_t)(kb + 1) * G,
                        lds_byte_off(&lA[nxt][0]), 0, G, G, (unsigned)K, (unsigned)K);
            tdm_load_2d(qw8 + (size_t)n0 * K + (size_t)(kb + 1) * G,
                        lds_byte_off(&lB[nxt][0]), 0, G, G, (unsigned)K, (unsigned)K);
        }
#elif defined(FILL_ASYNC)
        if (have_next) {
            const int nxt = cur ^ 1;
            const int frow = tid >> 1, fcol = (tid & 1) * 64;
            const uint8_t* gA = qx  + (size_t)(m0 + frow) * K + (size_t)(kb + 1) * G + fcol;
            const uint8_t* gB = qw8 + (size_t)(n0 + frow) * K + (size_t)(kb + 1) * G + fcol;
            uint8_t* sA = &lA[nxt][frow * G + fcol];
            uint8_t* sB = &lB[nxt][frow * G + fcol];
#pragma unroll
            for (int q = 0; q < 4; ++q) {
                async_cp16(gA + 16 * q, sA + 16 * q);
                async_cp16(gB + 16 * q, sB + 16 * q);
            }
        }
#else
        v4i pa[4], pb[4];
        const int frow = tid >> 1, fcol = (tid & 1) * 64;
        if (have_next) {
            const v4i* gA = (const v4i*)(qx  + (size_t)(m0 + frow) * K + (size_t)(kb + 1) * G + fcol);
            const v4i* gB = (const v4i*)(qw8 + (size_t)(n0 + frow) * K + (size_t)(kb + 1) * G + fcol);
#pragma unroll
            for (int q = 0; q < 4; ++q) { pa[q] = gA[q]; pb[q] = gB[q]; }
        }
#endif

        v16i af[2];
#pragma unroll
        for (int u = 0; u < 2; ++u)
            af[u] = lds_frag64(curA + arow0 + u * 16 * G);

        const float wsc = wscale[(size_t)nb * KB + kb];
        const float* sp = &lS[kb * G + srow0];
        float sc[2][8];
#pragma unroll
        for (int u = 0; u < 2; ++u) {
            float4 a0 = *(const float4*)(sp + u * 16);
            float4 a1 = *(const float4*)(sp + u * 16 + 4);
            sc[u][0] = a0.x * wsc; sc[u][1] = a0.y * wsc;
            sc[u][2] = a0.z * wsc; sc[u][3] = a0.w * wsc;
            sc[u][4] = a1.x * wsc; sc[u][5] = a1.y * wsc;
            sc[u][6] = a1.z * wsc; sc[u][7] = a1.w * wsc;
        }

        v16i bf = lds_frag64(curB + brow0);
#pragma unroll
        for (int t = 0; t < 4; ++t) {
            v16i bf_next;
            if (t < 3) bf_next = lds_frag64(curB + brow0 + (t + 1) * 16 * G);
#pragma unroll
            for (int u = 0; u < 2; ++u) {
                v8f p = __builtin_amdgcn_wmma_f32_16x16x128_fp8_fp8(
                            af[u], bf, (short)0, vzero, false, false);
#pragma unroll
                for (int r = 0; r < 8; ++r)
                    acc[u][t][r] += sc[u][r] * p[r];
            }
            if (t < 3) bf = bf_next;
        }

#if defined(FILL_TDM)
        __builtin_amdgcn_s_wait_tensorcnt(0);
        __syncthreads();
#elif defined(FILL_ASYNC)
        async_wait0();
        __syncthreads();
#else
        __syncthreads();
        if (have_next) {
            v4i* sA = (v4i*)&lA[cur ^ 1][frow * G + fcol];
            v4i* sB = (v4i*)&lB[cur ^ 1][frow * G + fcol];
#pragma unroll
            for (int q = 0; q < 4; ++q) { sA[q] = pa[q]; sB[q] = pb[q]; }
        }
        __syncthreads();
#endif
    }

#pragma unroll
    for (int t = 0; t < 4; ++t) {
        const int n = n0 + wn * 64 + t * 16 + lan;
        const float b = bias[n];
#pragma unroll
        for (int u = 0; u < 2; ++u)
#pragma unroll
            for (int r = 0; r < 8; ++r) {
                const int m = m0 + wm * 32 + u * 16 + r + 8 * hi;
                out[(size_t)m * N + n] = acc[u][t][r] + b;
            }
    }
}


extern "C" void kernel_launch(void* const* d_in, const int* in_sizes, int n_in,
                              void* d_out, int out_size, void* d_ws, size_t ws_size,
                              hipStream_t stream) {
    const float* x     = (const float*)d_in[0];
    const float* wq    = (const float*)d_in[1];
    const float* wscal = (const float*)d_in[2];
    const float* bias  = (const float*)d_in[3];
    float* out = (float*)d_out;

    const int N  = in_sizes[3];
    const int K  = (int)((long long)in_sizes[1] / N);
    const int M  = (int)((long long)in_sizes[0] / K);
    const int KB = K / G;

    uint8_t* qx  = (uint8_t*)d_ws;
    float*   sxT = (float*)(qx + (size_t)M * K);
    uint8_t* qw8 = (uint8_t*)((uint8_t*)sxT + (size_t)M * KB * 4);

    {
        int groups = M * KB;
        quant_act_kernel<<<groups / 8, 256, 0, stream>>>(x, qx, sxT, M, K, KB);
    }
    {
        size_t total4 = (size_t)N * K / 4;
        int blocks = (int)((total4 + 255) / 256);
        quant_wt_kernel<<<blocks, 256, 0, stream>>>(wq, qw8, total4);
    }
    {
        dim3 grid(N / G, M / G);
        gemm_fp8_kernel<<<grid, 256, 0, stream>>>(qx, sxT, qw8, wscal, bias, out,
                                                  M, N, K, KB);
    }
}
